// GATConv_65592740544797
// MI455X (gfx1250) — matrix operand order repaired and checked by reading, not compiled, not run on hardware
//
#include <hip/hip_runtime.h>

// ---------------------------------------------------------------------------
// CDNA5 (gfx1250) bf16 WMMA implementation of masked-MHA "GATConv" pipeline.
// All GEMMs use v_wmma_f32_16x16x32_bf16 (wave32, 16x16 tile per wave).
// V tiles in the flash-attention loop are staged to LDS with the Tensor Data
// Mover (tensor_load_to_lds + s_wait_tensorcnt) when the builtin is available.
// ---------------------------------------------------------------------------

typedef __attribute__((ext_vector_type(16))) __bf16       v16bf;
typedef __attribute__((ext_vector_type(8)))  __bf16       v8bf;
typedef __attribute__((ext_vector_type(8)))  float        v8f;
typedef __attribute__((ext_vector_type(4)))  unsigned int v4u;
typedef __attribute__((ext_vector_type(8)))  int          v8i;
typedef __attribute__((ext_vector_type(4)))  int          v4i;

union BF16x16 { v16bf v; v8bf h[2]; __bf16 e[16]; };
union BF16x8  { v8bf  v; __bf16 e[8]; };
union F32x8   { v8f   v; float  e[8]; };

#define BATCH 8
#define NTOK  1024
#define EDIM  256
#define HEADS 8
#define HD    32
#define QSCALE 0.17677669529663687f   // 1/sqrt(32)

#if defined(__gfx1250__) && __has_builtin(__builtin_amdgcn_tensor_load_to_lds)
#define HAVE_TDM 1
#else
#define HAVE_TDM 0
#endif

// Load a 16-element bf16 fragment as two 16-byte chunks (A or B operand).
__device__ inline v16bf load_frag(const __bf16* p0, const __bf16* p1) {
  BF16x16 f;
  f.h[0] = *(const v8bf*)p0;
  f.h[1] = *(const v8bf*)p1;
  return f.v;
}

// Strided 16-element bf16 fragment gather (for row-major LDS tiles).
__device__ inline v16bf load_frag_strided(const __bf16* p, int stride) {
  BF16x16 f;
#pragma unroll
  for (int e = 0; e < 16; ++e) f.e[e] = p[(size_t)e * stride];
  return f.v;
}

__device__ inline v8f wmma_bf16(v16bf a, v16bf b, v8f c) {
  // (neg_a, A, neg_b, B, c_mod, C, reuse_a, reuse_b)
  return __builtin_amdgcn_wmma_f32_16x16x32_bf16(false, a, false, b,
                                                 (short)0, c, false, false);
}

#if HAVE_TDM
// Issue a TDM 2D tile load: tile1 rows x tile0 elements of 2-byte data from
// global (row stride = stride0 elements) into LDS at lds_addr (bytes).
// D# layout per CDNA5 ISA ch.8 (group0 128b, group1 256b; groups 2/3 unused
// for 2D tensors -> zero).
__device__ inline void tdm_load_2d_bf16(unsigned int lds_addr,
                                        unsigned long long gaddr,
                                        unsigned int dim0, unsigned int dim1,
                                        unsigned int tile0, unsigned int tile1,
                                        unsigned int stride0) {
  v4u g0;
  g0.x = 1u;                                   // count=1 (valid), load, user
  g0.y = lds_addr;                             // lds_addr [63:32]
  g0.z = (unsigned int)gaddr;                  // global_addr lo
  g0.w = (unsigned int)((gaddr >> 32) & 0x01FFFFFFull) | 0x80000000u; // type=2
  v8i g1;
  g1[0] = (int)(1u << 16);                     // workgroup_mask=0, data_size=2B
  g1[1] = (int)((dim0 & 0xFFFFu) << 16);       // tensor_dim0 lo16 @ [63:48]
  g1[2] = (int)(((dim0 >> 16) & 0xFFFFu) | ((dim1 & 0xFFFFu) << 16));
  g1[3] = (int)(((dim1 >> 16) & 0xFFFFu) | ((tile0 & 0xFFFFu) << 16));
  g1[4] = (int)(tile1 & 0xFFFFu);              // tile_dim1, tile_dim2=0
  g1[5] = (int)stride0;                        // tensor_dim0_stride lo32
  g1[6] = 0;
  g1[7] = 0;
  v4i z4 = {0, 0, 0, 0};
#if __has_include(<hip/amd_detail/amd_gfx1250_TDM.h>)
  v8i z8 = {0, 0, 0, 0, 0, 0, 0, 0};
  __builtin_amdgcn_tensor_load_to_lds(g0, g1, z4, z4, z8, 0);   // 6-arg lane
#else
  __builtin_amdgcn_tensor_load_to_lds(g0, g1, z4, z4, 0);       // 5-arg lane
#endif
}
#endif

// ---------------------------------------------------------------------------
// fp32 -> bf16 conversion (activations + weights), one-shot
// ---------------------------------------------------------------------------
__global__ void cvt_kernel(const float* __restrict__ src,
                           __bf16* __restrict__ dst, int n) {
  int i = blockIdx.x * blockDim.x + threadIdx.x;
  if (i < n) dst[i] = (__bf16)src[i];
}

// ---------------------------------------------------------------------------
// QKV projection: qkv = x @ w_in^T + b_in, scattered to head-major q/k/v.
// One wave per 16x16 tile of the [8192 x 768] output.
// ---------------------------------------------------------------------------
__global__ __launch_bounds__(256) void qkv_kernel(
    const __bf16* __restrict__ xb, const __bf16* __restrict__ w_inb,
    const float* __restrict__ b_in,
    __bf16* __restrict__ q, __bf16* __restrict__ k, __bf16* __restrict__ v) {
  const int lane = threadIdx.x & 31;
  const int wave = threadIdx.x >> 5;
  const int hl = lane >> 4;          // which 16-lane half
  const int ln = lane & 15;
  const int NT = 768 / 16;           // 48 col tiles
  int tile = blockIdx.x * 8 + wave;
  int mt = tile / NT, nt = tile % NT;
  int c0 = nt * 16;

  const __bf16* arow = xb    + (size_t)(mt * 16 + ln) * EDIM;
  const __bf16* brow = w_inb + (size_t)(c0 + ln) * EDIM;  // B[k][c] = w_in[c][k]

  v8f acc = {};
#pragma unroll
  for (int ks = 0; ks < EDIM; ks += 32) {
    v16bf a  = load_frag(arow + ks + hl * 16, arow + ks + hl * 16 + 8);
    v16bf bb = load_frag(brow + ks + hl * 16, brow + ks + hl * 16 + 8);
    acc = wmma_bf16(a, bb, acc);
  }
  F32x8 d; d.v = acc;

  int col  = c0 + ln;              // one output column per lane
  float bias = b_in[col];
  int part = col >> 8;             // 0=q 1=k 2=v
  int e    = col & 255;
  int h    = e >> 5;
  int dcol = e & 31;
  __bf16* dst   = (part == 0) ? q : ((part == 1) ? k : v);
  float   scale = (part == 0) ? QSCALE : 1.0f;

#pragma unroll
  for (int r = 0; r < 8; ++r) {
    int M = mt * 16 + r + 8 * hl;  // global row
    int b = M >> 10, n = M & 1023;
    dst[((size_t)((b * HEADS + h) * NTOK + n)) * HD + dcol] =
        (__bf16)((d.e[r] + bias) * scale);
  }
}

// ---------------------------------------------------------------------------
// Flash attention with adjacency mask. One wave per (b, h, 16-query-row tile).
// Key loop in chunks of 32: TDM V-tile stage + 2 score WMMAs + online softmax
// + 2 PV WMMAs.
// ---------------------------------------------------------------------------
__global__ __launch_bounds__(128) void attn_kernel(
    const __bf16* __restrict__ q, const __bf16* __restrict__ k,
    const __bf16* __restrict__ v, const int* __restrict__ adj,
    __bf16* __restrict__ ctx) {
  __shared__ __attribute__((aligned(16))) __bf16 smem[4 * (16 * 32 + 32 * 32)];
  const int lane = threadIdx.x & 31;
  const int wave = threadIdx.x >> 5;
  const int hl = lane >> 4;
  const int ln = lane & 15;
  const int tile = blockIdx.x * 4 + wave;
  const int b    = tile >> 9;        // / (H * 64)
  const int h    = (tile >> 6) & 7;
  const int row0 = (tile & 63) << 4;

  __bf16* P  = smem + wave * (16 * 32 + 32 * 32); // 16x32 probs (bf16)
  __bf16* VT = P + 16 * 32;                       // 32x32 V tile

  const __bf16* qbase = q + (size_t)((b * HEADS + h) * NTOK) * HD;
  const __bf16* kbase = k + (size_t)((b * HEADS + h) * NTOK) * HD;
  const __bf16* vbase = v + (size_t)((b * HEADS + h) * NTOK) * HD;

  // Query A-fragment: K-dim == HD == 32 exactly, load once and reuse.
  const __bf16* qrow = qbase + (size_t)(row0 + ln) * HD;
  v16bf qa = load_frag(qrow + hl * 16, qrow + hl * 16 + 8);

  float mrow[8], lrow[8];
  F32x8 acc0, acc1;
#pragma unroll
  for (int r = 0; r < 8; ++r) {
    mrow[r] = -3.0e38f; lrow[r] = 0.0f; acc0.e[r] = 0.0f; acc1.e[r] = 0.0f;
  }

  for (int kc = 0; kc < NTOK; kc += 32) {
    // Prefetch next chunk's K rows into cache (global_prefetch_b8).
    int knext = (kc + 32) & (NTOK - 1);
    __builtin_prefetch(kbase + (size_t)(knext + lane) * HD, 0, 3);

#if HAVE_TDM
    // Stage V chunk (32 keys x 32 dims, row-major) into LDS via the Tensor
    // Data Mover; wave-uniform descriptor in SGPRs.
    {
      unsigned int lds =
          (unsigned int)__builtin_amdgcn_readfirstlane(
              (int)(unsigned int)(unsigned long long)VT);
      unsigned long long g =
          (unsigned long long)vbase + (unsigned long long)kc * (HD * 2);
      unsigned int glo = (unsigned int)__builtin_amdgcn_readfirstlane(
          (int)(unsigned int)g);
      unsigned int ghi = (unsigned int)__builtin_amdgcn_readfirstlane(
          (int)(unsigned int)(g >> 32));
      tdm_load_2d_bf16(lds, ((unsigned long long)ghi << 32) | glo,
                       /*dim0=*/HD, /*dim1=*/NTOK,
                       /*tile0=*/HD, /*tile1=*/32, /*stride0=*/HD);
    }
#else
    // Fallback: lane owns key (kc+lane), scatters its row into LDS column
    // `lane` -> later B-fragment reads are contiguous b128.
    {
      const __bf16* vrow = vbase + (size_t)(kc + lane) * HD;
      const v8bf* vp = (const v8bf*)vrow;
#pragma unroll
      for (int j = 0; j < 4; ++j) {
        BF16x8 t; t.v = vp[j];
#pragma unroll
        for (int e2 = 0; e2 < 8; ++e2)
          VT[(j * 8 + e2) * 32 + lane] = t.e[e2];
      }
    }
#endif

    // Scores: S = q . k^T  (B[d][m] = k[m][d], contiguous along d)
    v8f zero = {};
    const __bf16* kr0 = kbase + (size_t)(kc + ln) * HD;
    const __bf16* kr1 = kbase + (size_t)(kc + 16 + ln) * HD;
    v16bf kb0 = load_frag(kr0 + hl * 16, kr0 + hl * 16 + 8);
    v16bf kb1 = load_frag(kr1 + hl * 16, kr1 + hl * 16 + 8);
    F32x8 S0, S1;
    S0.v = wmma_bf16(qa, kb0, zero);
    S1.v = wmma_bf16(qa, kb1, zero);

    // Mask + online softmax (rows live in 16-lane halves of the wave).
#pragma unroll
    for (int r = 0; r < 8; ++r) {
      int qr = row0 + r + 8 * hl;
      const int* arow = adj + (size_t)qr * NTOK + kc;
      float s0 = (arow[ln]      != 0) ? S0.e[r] : -1.0e9f;
      float s1 = (arow[16 + ln] != 0) ? S1.e[r] : -1.0e9f;
      float rmax = fmaxf(s0, s1);
#pragma unroll
      for (int m = 1; m < 16; m <<= 1) rmax = fmaxf(rmax, __shfl_xor(rmax, m, 32));
      float mnew = fmaxf(mrow[r], rmax);
      float corr = __expf(mrow[r] - mnew);
      float p0 = __expf(s0 - mnew);
      float p1 = __expf(s1 - mnew);
      float rsum = p0 + p1;
#pragma unroll
      for (int m = 1; m < 16; m <<= 1) rsum += __shfl_xor(rsum, m, 32);
      lrow[r] = lrow[r] * corr + rsum;
      mrow[r] = mnew;
      acc0.e[r] *= corr;
      acc1.e[r] *= corr;
      int prow = r + 8 * hl;
      P[prow * 32 + ln]      = (__bf16)p0;   // re-layout D -> A via LDS
      P[prow * 32 + 16 + ln] = (__bf16)p1;
    }

#if HAVE_TDM
    // Wait for TDM tile before consuming it from LDS.
    __builtin_amdgcn_s_wait_tensorcnt(0);
    asm volatile("" ::: "memory");
#endif

    // PV: A = P (16x32), B = V chunk (32 x 32), two 16-col subtiles.
    v16bf pa = load_frag(P + ln * 32 + hl * 16, P + ln * 32 + hl * 16 + 8);
#if HAVE_TDM
    // LDS holds V row-major [key][d]: gather B fragments with stride HD.
    v16bf vb0 = load_frag_strided(VT + (size_t)(hl * 16) * HD + ln, HD);
    v16bf vb1 = load_frag_strided(VT + (size_t)(hl * 16) * HD + 16 + ln, HD);
#else
    const __bf16* vt0 = VT + (size_t)ln * 32 + hl * 16;
    const __bf16* vt1 = VT + (size_t)(16 + ln) * 32 + hl * 16;
    v16bf vb0 = load_frag(vt0, vt0 + 8);
    v16bf vb1 = load_frag(vt1, vt1 + 8);
#endif
    acc0.v = wmma_bf16(pa, vb0, acc0.v);
    acc1.v = wmma_bf16(pa, vb1, acc1.v);
  }

  // Normalize and write ctx in [B, N, E] layout (E = h*32 + d).
  __bf16* cbase = ctx + (size_t)(b * NTOK + row0) * EDIM + h * HD;
#pragma unroll
  for (int r = 0; r < 8; ++r) {
    float inv = 1.0f / lrow[r];
    int row = r + 8 * hl;
    cbase[(size_t)row * EDIM + ln]      = (__bf16)(acc0.e[r] * inv);
    cbase[(size_t)row * EDIM + 16 + ln] = (__bf16)(acc1.e[r] * inv);
  }
}

// ---------------------------------------------------------------------------
// Fused epilogue: out = leaky_relu((ctx @ w_out^T + b_out) @ w_lin^T + b_lin)
// One block = 16 output rows; phase 1 fills a 16x256 LDS stripe, phase 2
// consumes it as the A operand of the second GEMM.
// ---------------------------------------------------------------------------
__global__ __launch_bounds__(256) void out_kernel(
    const __bf16* __restrict__ ctx, const __bf16* __restrict__ w_outb,
    const float* __restrict__ b_out, const __bf16* __restrict__ w_linb,
    const float* __restrict__ b_lin, float* __restrict__ out) {
  __shared__ __attribute__((aligned(16))) __bf16 att[16 * 256];
  const int lane = threadIdx.x & 31;
  const int wave = threadIdx.x >> 5;
  const int hl = lane >> 4;
  const int ln = lane & 15;
  const int row0 = blockIdx.x * 16;

  // Phase 1: attended = ctx @ w_out^T + b_out  -> LDS (bf16)
#pragma unroll
  for (int t = 0; t < 2; ++t) {
    int c0 = wave * 32 + t * 16;
    const __bf16* ar = ctx    + (size_t)(row0 + ln) * EDIM;
    const __bf16* br = w_outb + (size_t)(c0 + ln) * EDIM;
    v8f acc = {};
#pragma unroll
    for (int ks = 0; ks < EDIM; ks += 32) {
      v16bf a  = load_frag(ar + ks + hl * 16, ar + ks + hl * 16 + 8);
      v16bf bb = load_frag(br + ks + hl * 16, br + ks + hl * 16 + 8);
      acc = wmma_bf16(a, bb, acc);
    }
    F32x8 d; d.v = acc;
    float bias = b_out[c0 + ln];
#pragma unroll
    for (int r = 0; r < 8; ++r)
      att[(r + 8 * hl) * 256 + c0 + ln] = (__bf16)(d.e[r] + bias);
  }
  __syncthreads();

  // Phase 2: out = attended @ w_lin^T + b_lin, then leaky_relu(0.2)
#pragma unroll
  for (int t = 0; t < 2; ++t) {
    int c0 = wave * 32 + t * 16;
    const __bf16* ar = att    + (size_t)ln * 256;
    const __bf16* br = w_linb + (size_t)(c0 + ln) * EDIM;
    v8f acc = {};
#pragma unroll
    for (int ks = 0; ks < EDIM; ks += 32) {
      v16bf a  = load_frag(ar + ks + hl * 16, ar + ks + hl * 16 + 8);
      v16bf bb = load_frag(br + ks + hl * 16, br + ks + hl * 16 + 8);
      acc = wmma_bf16(a, bb, acc);
    }
    F32x8 d; d.v = acc;
    float bias = b_lin[c0 + ln];
#pragma unroll
    for (int r = 0; r < 8; ++r) {
      float vv = d.e[r] + bias;
      vv = (vv >= 0.0f) ? vv : 0.2f * vv;
      out[(size_t)(row0 + r + 8 * hl) * EDIM + c0 + ln] = vv;
    }
  }
}

// ---------------------------------------------------------------------------
// Host launcher
// ---------------------------------------------------------------------------
extern "C" void kernel_launch(void* const* d_in, const int* in_sizes, int n_in,
                              void* d_out, int out_size, void* d_ws, size_t ws_size,
                              hipStream_t stream) {
  const float* x     = (const float*)d_in[0];
  const int*   adj   = (const int*)d_in[1];
  const float* w_in  = (const float*)d_in[2];
  const float* b_in  = (const float*)d_in[3];
  const float* w_out = (const float*)d_in[4];
  const float* b_out = (const float*)d_in[5];
  const float* w_lin = (const float*)d_in[6];
  const float* b_lin = (const float*)d_in[7];
  float* out = (float*)d_out;

  char* ws = (char*)d_ws;
  __bf16* xb     = (__bf16*)(ws);               // 8192*256      -> 4 MiB
  __bf16* w_inb  = (__bf16*)(ws + 4194304);     // 768*256
  __bf16* w_outb = (__bf16*)(ws + 4587520);     // 256*256
  __bf16* w_linb = (__bf16*)(ws + 4718592);     // 256*256
  __bf16* qd     = (__bf16*)(ws + 4849664);     // [B,H,N,D]     -> 4 MiB
  __bf16* kd     = (__bf16*)(ws + 9043968);     // 4 MiB
  __bf16* vd     = (__bf16*)(ws + 13238272);    // 4 MiB
  __bf16* ctx    = (__bf16*)(ws + 17432576);    // [B,N,E]       -> 4 MiB

  cvt_kernel<<<(2097152 + 255) / 256, 256, 0, stream>>>(x, xb, 2097152);
  cvt_kernel<<<(196608  + 255) / 256, 256, 0, stream>>>(w_in,  w_inb, 196608);
  cvt_kernel<<<(65536   + 255) / 256, 256, 0, stream>>>(w_out, w_outb, 65536);
  cvt_kernel<<<(65536   + 255) / 256, 256, 0, stream>>>(w_lin, w_linb, 65536);

  // 8192/16 row tiles * 768/16 col tiles = 24576 waves / 8 per block
  qkv_kernel<<<3072, 256, 0, stream>>>(xb, w_inb, b_in, qd, kd, vd);
  // B*H*(N/16) = 4096 waves / 4 per block
  attn_kernel<<<1024, 128, 0, stream>>>(qd, kd, vd, adj, ctx);
  // 8192/16 row stripes
  out_kernel<<<512, 256, 0, stream>>>(ctx, w_outb, b_out, w_linb, b_lin, out);
}
